// SpGraphAttentionLayer_652835029006
// MI455X (gfx1250) — matrix operand order repaired and checked by reading, not compiled, not run on hardware
//
#include <hip/hip_runtime.h>
#include <hip/hip_bf16.h>
#include <math.h>

#define N_NODES 50000
#define IN_F    256
#define HID_F   128
#define HEADS_N 4
#define OUT_F   512          // HEADS*HID
#define E_EDGES 800000
#define ALPHA_S 0.2f
#define EPS_S   1e-6f
// SCALE = sqrt(512)
#define INV_SCALE 0.044194173824159216f

typedef __attribute__((ext_vector_type(16))) _Float16 v16h;
typedef __attribute__((ext_vector_type(8)))  _Float16 v8h;
typedef __attribute__((ext_vector_type(8)))  float    v8f;

// ---------------- prep kernels ----------------

__global__ __launch_bounds__(256) void gat_conv_emb(const float* __restrict__ e,
                                                    _Float16* __restrict__ o) {
  int i = blockIdx.x * 256 + threadIdx.x;
  if (i >= N_NODES * IN_F) return;
  float v = e[i];
  if (i < IN_F) v = 0.0f;          // padding_idx = 0 -> zero row 0
  o[i] = (_Float16)v;
}

__global__ __launch_bounds__(256) void gat_conv_w(const float* __restrict__ w,
                                                  _Float16* __restrict__ o) {
  int i = blockIdx.x * 256 + threadIdx.x;
  if (i < HEADS_N * HID_F * IN_F) o[i] = (_Float16)w[i];
}

__global__ __launch_bounds__(256) void gat_zero_f32(float* __restrict__ p, int n) {
  int i = blockIdx.x * 256 + threadIdx.x;
  if (i < n) p[i] = 0.0f;
}

// ---------------- WMMA GEMM: X[N,512] = emb[N,256] @ W[512,256]^T + bias ----------------
// Block: 256 threads = 8 waves. Block b -> rows 16b..16b+15. Wave w -> cols 64w..64w+63.
// Each wave holds one A fragment (16x32 f16) and streams 4 B fragments per K-step.

__global__ __launch_bounds__(256) void gat_gemm_wmma(const _Float16* __restrict__ A,
                                                     const _Float16* __restrict__ B,
                                                     const float* __restrict__ bias,
                                                     float* __restrict__ X) {
  const int lane = threadIdx.x & 31;
  const int wave = threadIdx.x >> 5;
  const int row0 = blockIdx.x << 4;
  const int col0 = wave << 6;
  const int lr = lane & 15;     // A: tile row, B/D: tile col
  const int hi = lane >> 4;     // 0: K 0-7,16-23; 1: K 8-15,24-31

  v8f acc[4] = {v8f{0,0,0,0,0,0,0,0}, v8f{0,0,0,0,0,0,0,0},
                v8f{0,0,0,0,0,0,0,0}, v8f{0,0,0,0,0,0,0,0}};

  const _Float16* arow = A + (size_t)(row0 + lr) * IN_F;

  #pragma unroll
  for (int k0 = 0; k0 < IN_F; k0 += 32) {
    // A fragment 16x32: lane group hi holds K = k0+8*hi+{0..7} (elems 0-7)
    // and K = k0+8*hi+16+{0..7} (elems 8-15)  -> two b128 loads
    const int ka = k0 + hi * 8;
    v8h alo = *(const v8h*)(arow + ka);
    v8h ahi = *(const v8h*)(arow + ka + 16);
    v16h av = __builtin_shufflevector(alo, ahi, 0,1,2,3,4,5,6,7,8,9,10,11,12,13,14,15);
    #pragma unroll
    for (int t = 0; t < 4; ++t) {
      // B fragment 32x16: lane holds col gcol, K = k0+8*hi+{0..7} and k0+8*hi+16+{0..7}
      const int gcol = col0 + t * 16 + lr;       // flat col in [0,512) == W row
      const _Float16* brow = B + (size_t)gcol * IN_F + k0 + hi * 8;
      v8h blo = *(const v8h*)(brow);
      v8h bhi = *(const v8h*)(brow + 16);
      v16h bv = __builtin_shufflevector(blo, bhi, 0,1,2,3,4,5,6,7,8,9,10,11,12,13,14,15);
      acc[t] = __builtin_amdgcn_wmma_f32_16x16x32_f16(
          /*neg_a=*/false, av, /*neg_b=*/false, bv,
          /*c_mod=*/(short)0, acc[t], /*reuse_a=*/false, /*reuse_b=*/false);
    }
  }

  // D layout: lane lr = col; VGPR r -> row = 8*hi + r
  #pragma unroll
  for (int t = 0; t < 4; ++t) {
    const int gcol = col0 + t * 16 + lr;
    const float bv = bias[gcol];               // b is [4,128] flat == [512]
    #pragma unroll
    for (int r = 0; r < 8; ++r) {
      const int grow = row0 + hi * 8 + r;
      X[(size_t)grow * OUT_F + gcol] = acc[t][r] + bv;
    }
  }
}

// ---------------- attention projections: s1/s2 per (node, head) ----------------

__global__ __launch_bounds__(256) void gat_scores(const float* __restrict__ X,
                                                  const float* __restrict__ a_attn,
                                                  float* __restrict__ s1,
                                                  float* __restrict__ s2) {
  int idx = blockIdx.x * 256 + threadIdx.x;      // node*HEADS + h
  if (idx >= N_NODES * HEADS_N) return;
  const int node = idx >> 2;
  const int h = idx & 3;
  const float* xr = X + (size_t)node * OUT_F + h * HID_F;
  const float* a1 = a_attn + h * (2 * HID_F);
  const float* a2 = a1 + HID_F;
  float t1 = 0.0f, t2 = 0.0f;
  #pragma unroll 4
  for (int c = 0; c < HID_F; ++c) {
    float v = xr[c];
    t1 += v * a1[c];
    t2 += v * a2[c];
  }
  s1[idx] = t1;
  s2[idx] = t2;
}

// ---------------- edge scatter: one wave per (edge, head) ----------------

__global__ __launch_bounds__(256) void gat_edges(const int* __restrict__ edge,
                                                 const float* __restrict__ X,
                                                 const float* __restrict__ s1,
                                                 const float* __restrict__ s2,
                                                 float* __restrict__ agg,
                                                 float* __restrict__ rowsum) {
  const int tid = blockIdx.x * 256 + threadIdx.x;
  const int gw = tid >> 5;                       // global wave id
  const int lane = tid & 31;
  if (gw >= E_EDGES * HEADS_N) return;
  const int e = gw >> 2;
  const int h = gw & 3;
  const int src = edge[e];
  const int dst = edge[E_EDGES + e];
  float sc = s1[src * HEADS_N + h] + s2[dst * HEADS_N + h];
  sc = (sc >= 0.0f ? sc : ALPHA_S * sc) * INV_SCALE;
  const float w = expf(sc);
  if (lane == 0) atomicAdd(rowsum + src * HEADS_N + h, w);
  const float* xd = X + (size_t)dst * OUT_F + h * HID_F;
  float* ag = agg + (size_t)src * OUT_F + h * HID_F;
  #pragma unroll
  for (int c = lane; c < HID_F; c += 32) {
    atomicAdd(ag + c, w * xd[c]);
  }
}

// ---------------- finalize: divide, LayerNorm (ddof=1, std+eps), ELU ----------------

__global__ __launch_bounds__(256) void gat_finalize(const float* __restrict__ agg,
                                                    const float* __restrict__ rowsum,
                                                    const float* __restrict__ gain,
                                                    const float* __restrict__ bias,
                                                    float* __restrict__ out) {
  const int node = blockIdx.x;
  const int lane = threadIdx.x & 31;
  const int wave = threadIdx.x >> 5;

  float v[2];
  #pragma unroll
  for (int i = 0; i < 2; ++i) {
    const int col = threadIdx.x + i * 256;
    const int h = col >> 7;
    float rs = rowsum[node * HEADS_N + h];
    float denom = (rs == 0.0f) ? 1.0f : rs;
    v[i] = agg[(size_t)node * OUT_F + col] / denom;
  }

  float s = v[0] + v[1];
  float q = v[0] * v[0] + v[1] * v[1];
  #pragma unroll
  for (int o = 16; o > 0; o >>= 1) {
    s += __shfl_xor(s, o, 32);
    q += __shfl_xor(q, o, 32);
  }

  __shared__ float sh_s[8], sh_q[8];
  __shared__ float sh_mean, sh_rstd;
  if (lane == 0) { sh_s[wave] = s; sh_q[wave] = q; }
  __syncthreads();
  if (threadIdx.x == 0) {
    float S = 0.0f, Q = 0.0f;
    #pragma unroll
    for (int i = 0; i < 8; ++i) { S += sh_s[i]; Q += sh_q[i]; }
    float mean = S * (1.0f / OUT_F);
    float var = (Q - (float)OUT_F * mean * mean) * (1.0f / (OUT_F - 1));
    var = var < 0.0f ? 0.0f : var;
    sh_mean = mean;
    sh_rstd = 1.0f / (sqrtf(var) + EPS_S);
  }
  __syncthreads();
  const float mean = sh_mean;
  const float rstd = sh_rstd;
  #pragma unroll
  for (int i = 0; i < 2; ++i) {
    const int col = threadIdx.x + i * 256;
    float nv = gain[col] * (v[i] - mean) * rstd + bias[col];
    out[(size_t)node * OUT_F + col] = nv > 0.0f ? nv : expm1f(nv);
  }
}

// ---------------- host launcher ----------------

extern "C" void kernel_launch(void* const* d_in, const int* in_sizes, int n_in,
                              void* d_out, int out_size, void* d_ws, size_t ws_size,
                              hipStream_t stream) {
  // inputs: data, edge, embed_table, W, b, a_attn, ln_gain, ln_bias
  const int*   edge    = (const int*)d_in[1];
  const float* embed   = (const float*)d_in[2];
  const float* W       = (const float*)d_in[3];
  const float* b       = (const float*)d_in[4];
  const float* a_attn  = (const float*)d_in[5];
  const float* ln_gain = (const float*)d_in[6];
  const float* ln_bias = (const float*)d_in[7];
  float* out = (float*)d_out;

  char* ws = (char*)d_ws;
  size_t off = 0;
  _Float16* embf16 = (_Float16*)(ws + off); off += (size_t)N_NODES * IN_F * 2;          // 25.6 MB
  _Float16* Wf16   = (_Float16*)(ws + off); off += (size_t)HEADS_N * HID_F * IN_F * 2;  // 256 KB
  float* X       = (float*)(ws + off); off += (size_t)N_NODES * OUT_F * 4;              // 102.4 MB
  float* agg     = (float*)(ws + off); off += (size_t)N_NODES * OUT_F * 4;              // 102.4 MB
  float* rowsum  = (float*)(ws + off); off += (size_t)N_NODES * HEADS_N * 4;            // 0.8 MB (adjacent to agg)
  float* s1      = (float*)(ws + off); off += (size_t)N_NODES * HEADS_N * 4;
  float* s2      = (float*)(ws + off); off += (size_t)N_NODES * HEADS_N * 4;

  // 1) conversions + zero accumulators (agg+rowsum are contiguous -> one zero pass)
  {
    int n = N_NODES * IN_F;
    gat_conv_emb<<<(n + 255) / 256, 256, 0, stream>>>(embed, embf16);
  }
  {
    int n = HEADS_N * HID_F * IN_F;
    gat_conv_w<<<(n + 255) / 256, 256, 0, stream>>>(W, Wf16);
  }
  {
    int n = N_NODES * OUT_F + N_NODES * HEADS_N;
    gat_zero_f32<<<(n + 255) / 256, 256, 0, stream>>>(agg, n);
  }

  // 2) WMMA GEMM: 3125 blocks (16 rows each), 8 waves covering 512 cols
  gat_gemm_wmma<<<N_NODES / 16, 256, 0, stream>>>(embf16, Wf16, b, X);

  // 3) attention projections
  {
    int n = N_NODES * HEADS_N;
    gat_scores<<<(n + 255) / 256, 256, 0, stream>>>(X, a_attn, s1, s2);
  }

  // 4) edge scatter: one wave per (edge, head)
  {
    long long waves = (long long)E_EDGES * HEADS_N;          // 3.2M waves
    long long thr = waves * 32;                              // 102.4M threads
    int blocks = (int)((thr + 255) / 256);                   // 400000
    gat_edges<<<blocks, 256, 0, stream>>>(edge, X, s1, s2, agg, rowsum);
  }

  // 5) finalize
  gat_finalize<<<N_NODES, 256, 0, stream>>>(agg, rowsum, ln_gain, ln_bias, out);
}
